// XQHNet_18107582120336
// MI455X (gfx1250) — matrix operand order repaired and checked by reading, not compiled, not run on hardware
//
#include <hip/hip_runtime.h>

// ---------------- problem constants (match reference) ----------------
#define N_NODES 10000
#define N_E     160000
#define N_EF    240000
#define C_      64
#define H_      32
#define O_      16
#define B_      32
#define ND_     128
#define CUTOFF_ 5.0f

static_assert(N_NODES % 16 == 0, "");
static_assert(N_EF % 16 == 0, "");
static_assert(N_E % 4 == 0, "");

// l of spherical component 0..8 -> {0,1,1,1,2,2,2,2,2}
__device__ __forceinline__ int lc_of(int i) { return i == 0 ? 0 : (i < 4 ? 1 : 2); }

// ---------------- CDNA5 WMMA (wave32) helpers ----------------
typedef __bf16 bf16;
typedef bf16  v16bf __attribute__((ext_vector_type(16)));
typedef float v8f   __attribute__((ext_vector_type(8)));

__device__ __forceinline__ v8f wmma_bf16(v16bf a, v16bf b, v8f c) {
  // 8 args: (neg_a, A, neg_b, B, c_mod, C, reuse_a, reuse_b)
  return __builtin_amdgcn_wmma_f32_16x16x32_bf16(false, a, false, b, (short)0, c,
                                                 false, false);
}

// A-matrix (16x32, M x K) element K index for lane/half-j  (ISA 7.12.2):
// per lane this walks K in two CONTIGUOUS runs of 8 -> unit-stride A gives b128 loads.
__device__ __forceinline__ int ka_idx(int lane, int j) {
  return ((j & 8) << 1) | ((lane & 16) >> 1) | (j & 7);
}
// B-matrix (32x16, K x N) element K index for lane/half-j
__device__ __forceinline__ int kb_idx(int lane, int j) {
  return ((j & 8) << 1) | ((lane & 16) >> 1) | (j & 7);
}

// One 16x16 tile with pre-swizzled bf16 B fragments.
// A[m][k] = A[m*a_ms + k*a_ks] (pointer pre-offset to tile origin).
// Bf = fragment array for this (matrix, n-tile): Bf[kt*32 + lane] is a v16bf.
template <int KTILES>
__device__ __forceinline__ v8f gemm_tile_fb(const float* __restrict__ A, int a_ms, int a_ks,
                                            const v16bf* __restrict__ Bf, int lane, v8f acc) {
  const int m = lane & 15;
#pragma unroll
  for (int kt = 0; kt < KTILES; ++kt) {
    v16bf a;
#pragma unroll
    for (int j = 0; j < 16; ++j)
      a[j] = (bf16)A[m * a_ms + (kt * 32 + ka_idx(lane, j)) * a_ks];
    acc = wmma_bf16(a, Bf[kt * 32 + lane], acc);
  }
  return acc;
}

// ---------------- utility ----------------
__global__ void k_zero(float* p, long n) {
  long i = (long)blockIdx.x * blockDim.x + threadIdx.x;
  if (i < n) p[i] = 0.0f;
}

// Pre-swizzle a batch of row-major KxN f32 matrices into WMMA B fragments (bf16).
// Layout: dst[((mat*NT + nt)*KT + kt)*32 + lane] = v16bf fragment.
// grid = (K/32, N/16, nmat), block = 32.
__global__ void k_prep_b(const float* __restrict__ src, v16bf* __restrict__ dst,
                         int K, int Nn) {
  const int kt = blockIdx.x, nt = blockIdx.y, mat = blockIdx.z;
  const int KT = K >> 5, NT = Nn >> 4;
  const int lane = threadIdx.x & 31;
  const float* S = src + (long)mat * K * Nn;
  v16bf f;
#pragma unroll
  for (int j = 0; j < 16; ++j)
    f[j] = (bf16)S[(kt * 32 + kb_idx(lane, j)) * Nn + nt * 16 + (lane & 15)];
  dst[(((long)mat * NT + nt) * KT + kt) * 32 + lane] = f;
}

// ---------------- embedding: x0 = embed_table[at_no] @ embed_lin ----------------
// grid: N/16 blocks, block 128 (4 waves = 4 column tiles of C)
__global__ void k_embed(const int* __restrict__ at_no, const float* __restrict__ table,
                        const v16bf* __restrict__ elinF, float* __restrict__ x0,
                        float* __restrict__ nf) {
  const int wave = threadIdx.x >> 5, lane = threadIdx.x & 31;
  const int m0 = blockIdx.x * 16, c0 = wave * 16;
  const int m = lane & 15;
  const float* arow = table + (long)at_no[m0 + m] * ND_;  // gathered row (unit K stride)
  const v16bf* Bf = elinF + wave * 4 * 32;                // nt = wave, KT = 4
  v8f acc = {};
#pragma unroll
  for (int kt = 0; kt < ND_ / 32; ++kt) {
    v16bf a;
#pragma unroll
    for (int j = 0; j < 16; ++j) a[j] = (bf16)arow[kt * 32 + ka_idx(lane, j)];
    acc = wmma_bf16(a, Bf[kt * 32 + lane], acc);
  }
  const int n = lane & 15, mb = (lane >> 4) << 3;
#pragma unroll
  for (int r = 0; r < 8; ++r) {
    int node = m0 + mb + r;
    float v = acc[r];
    x0[node * C_ + c0 + n] = v;
    nf[(long)node * (9 * C_) + 0 * C_ + c0 + n] = v;  // component 0, (N,9,C) layout
  }
}

// ---------------- edge geometry: rbf (and rsh for conv edges) ----------------
__global__ void k_geom(const int* __restrict__ ei, int ne, const float* __restrict__ pos,
                       float* __restrict__ rbf, float* __restrict__ rsh) {
  int e = blockIdx.x * blockDim.x + threadIdx.x;
  if (e >= ne) return;
  int s = ei[e], d = ei[ne + e];
  float vx = pos[d * 3 + 0] - pos[s * 3 + 0];
  float vy = pos[d * 3 + 1] - pos[s * 3 + 1];
  float vz = pos[d * 3 + 2] - pos[s * 3 + 2];
  float r = sqrtf(vx * vx + vy * vy + vz * vz);
  if (rsh) {
    float inv = 1.0f / (r + 1e-8f);
    float x = vx * inv, y = vy * inv, z = vz * inv;
    float* o = rsh + (long)e * 9;
    o[0] = 1.0f; o[1] = x; o[2] = y; o[3] = z;
    o[4] = x * y; o[5] = y * z; o[6] = 3.0f * z * z - 1.0f;
    o[7] = x * z; o[8] = x * x - y * y;
  }
  float t = fminf(fmaxf(r / CUTOFF_, 0.0f), 1.0f);
  float env = 0.5f * (__cosf(3.14159265358979f * t) + 1.0f);
  const float width = CUTOFF_ / B_;
#pragma unroll
  for (int b = 0; b < B_; ++b) {
    float ctr = CUTOFF_ * (float)b / (float)(B_ - 1);
    float u = (r - ctr) / width;
    rbf[(long)e * B_ + b] = __expf(-0.5f * u * u) * env;
  }
}

// ---------------- conv message + scatter (per layer) ----------------
// block 256 = 4 edges x 64 channels; conv_w_rbf[k] (24KB) staged in LDS.
// (N,9,C) layout -> gathers and atomics fully coalesced per component.
__global__ void k_conv_msg(const int* __restrict__ ei, const float* __restrict__ rbf,
                           const float* __restrict__ rsh, const float* __restrict__ nf,
                           const float* __restrict__ wk, float* __restrict__ agg) {
  __shared__ float Ws[B_ * 3 * C_];  // 6144 floats
  __shared__ float rb[4][B_];
  __shared__ float sh[4][9];
  const int tid = threadIdx.x;
  const int e0 = blockIdx.x * 4;
  for (int i = tid; i < B_ * 3 * C_; i += 256) Ws[i] = wk[i];
  for (int i = tid; i < 4 * B_; i += 256) rb[i >> 5][i & 31] = rbf[(long)(e0 + (i >> 5)) * B_ + (i & 31)];
  if (tid < 36) sh[tid / 9][tid % 9] = rsh[(long)(e0 + tid / 9) * 9 + tid % 9];
  __syncthreads();
  const int el = tid >> 6, c = tid & 63;
  const int e = e0 + el;
  const int s = ei[e], d = ei[N_E + e];
  float w3[3] = {0.f, 0.f, 0.f};
#pragma unroll
  for (int b = 0; b < B_; ++b) {
    float v = rb[el][b];
    w3[0] += v * Ws[b * 192 + c];
    w3[1] += v * Ws[b * 192 + 64 + c];
    w3[2] += v * Ws[b * 192 + 128 + c];
  }
  const float* xj = nf + (long)s * (9 * C_) + c;
  float xj0 = xj[0];
  float* ag = agg + (long)d * (9 * C_) + c;
#pragma unroll
  for (int i = 0; i < 9; ++i) {
    float msg = w3[lc_of(i)] * (xj0 * sh[el][i] + xj[i * C_]);
    atomicAdd(&ag[i * C_], msg);  // L2-resident agg (23MB << 192MB L2), coalesced
  }
}

// ---------------- node update, component 0: silu + gate ----------------
__global__ void k_upd0(const float* __restrict__ agg, const v16bf* __restrict__ Ck,
                       float* __restrict__ nf, float* __restrict__ gate) {
  const int wave = threadIdx.x >> 5, lane = threadIdx.x & 31;
  const int m0 = blockIdx.x * 16, d0 = wave * 16;
  v8f acc = {};
  // A: agg[n][0][c] (unit K stride); B: cmix[k][0], n-tile = wave
  acc = gemm_tile_fb<2>(agg + (long)m0 * (9 * C_), 9 * C_, 1,
                        Ck + (wave * 2) * 32, lane, acc);
  const int n = lane & 15, mb = (lane >> 4) << 3;
#pragma unroll
  for (int r = 0; r < 8; ++r) {
    int node = m0 + mb + r, dc = d0 + n;
    float u = acc[r];
    float sg = 1.0f / (1.0f + __expf(-u));
    gate[node * C_ + dc] = sg;
    nf[(long)node * (9 * C_) + dc] += u * sg;  // silu, component 0
  }
}

// ---------------- node update, components 1..8: gated ----------------
__global__ void k_updL(const float* __restrict__ agg, const v16bf* __restrict__ Ck,
                       const float* __restrict__ gate, float* __restrict__ nf) {
  const int i = blockIdx.y + 1;
  const int wave = threadIdx.x >> 5, lane = threadIdx.x & 31;
  const int m0 = blockIdx.x * 16, d0 = wave * 16;
  const int l = lc_of(i);
  v8f acc = {};
  acc = gemm_tile_fb<2>(agg + (long)m0 * (9 * C_) + i * C_, 9 * C_, 1,
                        Ck + ((l * 4 + wave) * 2) * 32, lane, acc);
  const int n = lane & 15, mb = (lane >> 4) << 3;
#pragma unroll
  for (int r = 0; r < 8; ++r) {
    int node = m0 + mb + r, dc = d0 + n;
    nf[(long)node * (9 * C_) + i * C_ + dc] += acc[r] * gate[node * C_ + dc];
  }
}

// ---------------- projection h[n,i,h] = sum_c nf[n,i,c] * proj[lc[i]][c][h] ----------------
// grid N/16, block 256: 8 waves cover 9 comps x 2 H-tiles = 18 tasks
__global__ void k_proj(const float* __restrict__ nf, const v16bf* __restrict__ Pt,
                       float* __restrict__ hbuf) {
  const int wave = threadIdx.x >> 5, lane = threadIdx.x & 31;
  const int m0 = blockIdx.x * 16;
  for (int task = wave; task < 18; task += 8) {
    const int i = task >> 1, nt = task & 1, l = lc_of(i);
    v8f acc = {};
    acc = gemm_tile_fb<2>(nf + (long)m0 * (9 * C_) + i * C_, 9 * C_, 1,
                          Pt + ((l * 2 + nt) * 2) * 32, lane, acc);
    const int n = lane & 15, mb = (lane >> 4) << 3;
#pragma unroll
    for (int r = 0; r < 8; ++r) {
      int node = m0 + mb + r;
      hbuf[(long)node * (9 * H_) + i * H_ + nt * 16 + n] = acc[r];  // coalesced
    }
  }
}

// ---------------- diag accumulation: diag[n,o,i] += sum_h h[n,i,h]*odm[lc[i],h,o] ----------------
// 9 comps staged in LDS, then one contiguous 144-float-per-node writeback.
__global__ void k_diag(const float* __restrict__ hbuf, const v16bf* __restrict__ odmF,
                       float* __restrict__ diag) {
  __shared__ float Dt[16 * O_ * 9];  // 2304 floats
  const int tid = threadIdx.x;
  const int wave = tid >> 5, lane = tid & 31;
  const int m0 = blockIdx.x * 16;
  for (int i = wave; i < 9; i += 8) {
    v8f acc = {};
    acc = gemm_tile_fb<1>(hbuf + (long)m0 * (9 * H_) + i * H_, 9 * H_, 1,
                          odmF + lc_of(i) * 32, lane, acc);
    const int n = lane & 15, mb = (lane >> 4) << 3;
#pragma unroll
    for (int r = 0; r < 8; ++r) Dt[(mb + r) * (O_ * 9) + n * 9 + i] = acc[r];
  }
  __syncthreads();
  float* dbase = diag + (long)m0 * (O_ * 9);
  for (int idx = tid; idx < 16 * O_ * 9; idx += 256) dbase[idx] += Dt[idx];  // coalesced RMW
}

// ---------------- node-lin term: diag[n,o,0] += x0 @ out_node_lin ----------------
__global__ void k_nodelin(const float* __restrict__ x0, const v16bf* __restrict__ onlF,
                          float* __restrict__ diag) {
  const int lane = threadIdx.x & 31;
  const int m0 = blockIdx.x * 16;
  v8f acc = {};
  acc = gemm_tile_fb<2>(x0 + (long)m0 * C_, C_, 1, onlF, lane, acc);
  const int n = lane & 15, mb = (lane >> 4) << 3;
#pragma unroll
  for (int r = 0; r < 8; ++r) {
    int node = m0 + mb + r;
    diag[((long)node * O_ + n) * 9 + 0] += acc[r];
  }
}

// ---------------- full-edge kernel (per t): off[e,o,i] += sum_h we*(h_s+h_d)*oom ----------------
// block 256 handles 16 full edges; fij tile built in LDS, 9 WMMAs, contiguous writeback.
__global__ void k_off(const int* __restrict__ eif, const float* __restrict__ rbfF,
                      const float* __restrict__ hbuf, const float* __restrict__ twt,
                      const v16bf* __restrict__ oomF, float* __restrict__ off) {
  __shared__ float Tw[B_ * 3 * H_];      // 3072 floats (12KB)
  __shared__ float rb[16 * B_];          // 512
  __shared__ float we[16 * 3 * H_];      // 1536
  __shared__ float fij[16 * 9 * H_];     // 4608 (18KB)
  __shared__ float Ot[16 * O_ * 9];      // 2304 (9KB)
  __shared__ int   sd[2][16];
  const int tid = threadIdx.x;
  const int e0 = blockIdx.x * 16;
  for (int i = tid; i < B_ * 3 * H_; i += 256) Tw[i] = twt[i];
  for (int i = tid; i < 16 * B_; i += 256) rb[i] = rbfF[(long)e0 * B_ + i];
  if (tid < 32) {
    int node = eif[(long)(tid >> 4) * N_EF + e0 + (tid & 15)];
    sd[tid >> 4][tid & 15] = node;
    const float* row = hbuf + (long)node * (9 * H_);   // 1152B row
#pragma unroll
    for (int q = 0; q < 5; ++q) __builtin_prefetch(row + q * 64, 0, 1);  // global_prefetch_b8
  }
  __syncthreads();
  // we[e][j*H+h] = rbf_full[e] . trans_w_rbf[:, j*H+h]   (covers prefetch latency)
#pragma unroll
  for (int it = 0; it < 6; ++it) {
    int idx = tid + it * 256;                 // 16*96 = 1536 entries
    int e = idx / 96, jh = idx % 96;
    float acc = 0.f;
#pragma unroll
    for (int b = 0; b < B_; ++b) acc += rb[e * B_ + b] * Tw[b * 96 + jh];
    we[idx] = acc;
  }
  __syncthreads();
  // fij[e][i][h] = we[e][lc[i]][h] * (h[src]+h[dst])[i][h]  -- coalesced over h
  for (int it = 0; it < 18; ++it) {
    int idx = tid + it * 256;                 // 16*288 = 4608 entries
    int e = idx / 288, ih = idx % 288;
    int i = ih >> 5, hh = ih & 31;
    int s = sd[0][e], d = sd[1][e];
    float hs = hbuf[(long)s * (9 * H_) + ih] + hbuf[(long)d * (9 * H_) + ih];
    fij[idx] = we[e * 96 + lc_of(i) * H_ + hh] * hs;
  }
  __syncthreads();
  const int wave = tid >> 5, lane = tid & 31;
  for (int i = wave; i < 9; i += 8) {
    v8f acc = {};
    acc = gemm_tile_fb<1>(fij + i * H_, 9 * H_, 1, oomF + lc_of(i) * 32, lane, acc);
    const int n = lane & 15, mb = (lane >> 4) << 3;
#pragma unroll
    for (int r = 0; r < 8; ++r) Ot[(mb + r) * (O_ * 9) + n * 9 + i] = acc[r];
  }
  __syncthreads();
  float* obase = off + (long)e0 * (O_ * 9);
  for (int idx = tid; idx < 16 * O_ * 9; idx += 256) obase[idx] += Ot[idx];  // coalesced RMW
}

// ---------------- host launcher ----------------
extern "C" void kernel_launch(void* const* d_in, const int* in_sizes, int n_in,
                              void* d_out, int out_size, void* d_ws, size_t ws_size,
                              hipStream_t stream) {
  (void)in_sizes; (void)n_in; (void)ws_size;
  const int*   at_no = (const int*)  d_in[0];
  const float* pos   = (const float*)d_in[1];
  const int*   ei    = (const int*)  d_in[2];
  const int*   eif   = (const int*)  d_in[3];
  const float* table = (const float*)d_in[4];
  const float* elin  = (const float*)d_in[5];
  const float* cwrbf = (const float*)d_in[6];   // (3,B,3C)
  const float* cmix  = (const float*)d_in[7];   // (3,3,C,C)
  const float* tproj = (const float*)d_in[8];   // (2,3,C,H)
  const float* twrbf = (const float*)d_in[9];   // (2,B,3H)
  const float* odm   = (const float*)d_in[10];  // (3,H,O)
  const float* oom   = (const float*)d_in[11];  // (3,H,O)
  const float* onl   = (const float*)d_in[12];  // (C,O)

  float* out  = (float*)d_out;
  float* diag = out;                           // N*O*9
  float* off  = out + (long)N_NODES * O_ * 9;  // EF*O*9

  float* ws   = (float*)d_ws;
  float* x0   = ws; ws += (long)N_NODES * C_;
  float* nf   = ws; ws += (long)N_NODES * 9 * C_;   // (N,9,C)
  float* agg  = ws; ws += (long)N_NODES * 9 * C_;   // (N,9,C)
  float* gate = ws; ws += (long)N_NODES * C_;
  float* rbf  = ws; ws += (long)N_E * B_;
  float* rsh  = ws; ws += (long)N_E * 9;
  float* rbfF = ws; ws += (long)N_EF * B_;
  float* hbuf = ws; ws += (long)N_NODES * 9 * H_;   // (N,9,H)
  // pre-swizzled bf16 B-fragments (32B aligned: all offsets above are x8 floats)
  v16bf* fb    = (v16bf*)ws;
  v16bf* elinF = fb; fb += 1 * 4 * 4 * 32;  // ND x C  : KT=4 NT=4
  v16bf* cmixF = fb; fb += 9 * 4 * 2 * 32;  // C x C   : KT=2 NT=4, 9 mats
  v16bf* tprjF = fb; fb += 6 * 2 * 2 * 32;  // C x H   : KT=2 NT=2, 6 mats
  v16bf* odmF  = fb; fb += 3 * 1 * 1 * 32;  // H x O
  v16bf* oomF  = fb; fb += 3 * 1 * 1 * 32;  // H x O
  v16bf* onlF  = fb; fb += 1 * 1 * 2 * 32;  // C x O

  // weight pre-swizzle (tiny, once per launch)
  k_prep_b<<<dim3(4, 4, 1), 32, 0, stream>>>(elin, elinF, ND_, C_);
  k_prep_b<<<dim3(2, 4, 9), 32, 0, stream>>>(cmix, cmixF, C_, C_);
  k_prep_b<<<dim3(2, 2, 6), 32, 0, stream>>>(tproj, tprjF, C_, H_);
  k_prep_b<<<dim3(1, 1, 3), 32, 0, stream>>>(odm, odmF, H_, O_);
  k_prep_b<<<dim3(1, 1, 3), 32, 0, stream>>>(oom, oomF, H_, O_);
  k_prep_b<<<dim3(2, 1, 1), 32, 0, stream>>>(onl, onlF, C_, O_);

  const long n_out = (long)out_size;
  const long n_nf  = (long)N_NODES * 9 * C_;
  k_zero<<<(int)((n_out + 255) / 256), 256, 0, stream>>>(out, n_out);
  k_zero<<<(int)((n_nf + 255) / 256), 256, 0, stream>>>(nf, n_nf);

  k_embed<<<N_NODES / 16, 128, 0, stream>>>(at_no, table, elinF, x0, nf);
  k_geom<<<(N_E + 255) / 256, 256, 0, stream>>>(ei, N_E, pos, rbf, rsh);
  k_geom<<<(N_EF + 255) / 256, 256, 0, stream>>>(eif, N_EF, pos, rbfF, nullptr);

  for (int k = 0; k < 3; ++k) {
    k_zero<<<(int)((n_nf + 255) / 256), 256, 0, stream>>>(agg, n_nf);
    k_conv_msg<<<N_E / 4, 256, 0, stream>>>(ei, rbf, rsh, nf, cwrbf + (long)k * B_ * 3 * C_, agg);
    const v16bf* Ck = cmixF + (long)k * 3 * 4 * 2 * 32;  // 3 mats per layer
    k_upd0<<<N_NODES / 16, 128, 0, stream>>>(agg, Ck, nf, gate);
    k_updL<<<dim3(N_NODES / 16, 8), 128, 0, stream>>>(agg, Ck, gate, nf);
    if (k >= 1) {
      int t = k - 1;
      const v16bf* Pt = tprjF + (long)t * 3 * 2 * 2 * 32;
      k_proj<<<N_NODES / 16, 256, 0, stream>>>(nf, Pt, hbuf);
      k_diag<<<N_NODES / 16, 256, 0, stream>>>(hbuf, odmF, diag);
      k_off<<<N_EF / 16, 256, 0, stream>>>(eif, rbfF, hbuf, twrbf + (long)t * B_ * 3 * H_, oomF, off);
    }
  }
  k_nodelin<<<N_NODES / 16, 32, 0, stream>>>(x0, onlF, diag);
}
